// NoRGa_Attention_26491358282041
// MI455X (gfx1250) — compile-verified, matrix operand order checked by reading, not run on hardware
//
#include <hip/hip_runtime.h>
#include <hip/hip_bf16.h>

// ---------------------------------------------------------------------------
// NoRGa attention for MI455X (gfx1250), bf16 WMMA (v_wmma_f32_16x16x32_bf16)
// B=16, N=576, C=768, H=12, D=64, P=16.  Key sequence padded: 16+576+16 = 608.
// GEMMs use 32x32 register blocking: 4 WMMAs per 4 fragment loads.
// ---------------------------------------------------------------------------

typedef __bf16 bf16;
typedef __bf16 v16bf __attribute__((ext_vector_type(16)));
typedef float  v8f   __attribute__((ext_vector_type(8)));

#define BATCH 16
#define SEQ   576
#define CDIM  768
#define HEADS 12
#define DHEAD 64
#define PFX   16
#define KPAD  608              // 16 prefix + 576 + 16 zero-pad  (19 * 32)
#define TOK   (BATCH * SEQ)    // 9216

// Load a 16x32 bf16 A-fragment (ISA 7.12.2 16-bit A layout) from row-major
// memory.  p points at (row base + k0 + hi*8); elements j=0..7 at +0..7,
// j=8..15 at +16..23.  B-fragments use the mirrored load from B^T row-major.
static __device__ __forceinline__ v16bf load_frag(const bf16* p) {
    v16bf f;
    *reinterpret_cast<uint4*>(&f)       = *reinterpret_cast<const uint4*>(p);
    *(reinterpret_cast<uint4*>(&f) + 1) = *reinterpret_cast<const uint4*>(p + 16);
    return f;
}

static __device__ __forceinline__ v8f wmma_bf16(v16bf a, v16bf b, v8f c) {
    return __builtin_amdgcn_wmma_f32_16x16x32_bf16(false, a, false, b,
                                                   (short)0, c, false, false);
}

// --------------------------- fp32 -> bf16 cast (x4) -------------------------
__global__ void cast_bf16_kernel(const float* __restrict__ in,
                                 bf16* __restrict__ out, int n4) {
    int i = blockIdx.x * blockDim.x + threadIdx.x;
    if (i >= n4) return;
    float4 v = *reinterpret_cast<const float4*>(in + (size_t)i * 4);
    bf16 t[4] = { (bf16)v.x, (bf16)v.y, (bf16)v.z, (bf16)v.w };
    *reinterpret_cast<uint2*>(out + (size_t)i * 4) = *reinterpret_cast<uint2*>(t);
}

// ------------------- scatter prompt prefix into k_full / v_t ----------------
// prompt: [B,2,P,H,D] fp32.  k_full: [B,H,KPAD,D].  v_t: [B,H,D,KPAD].
__global__ void prefix_kernel(const float* __restrict__ prompt,
                              bf16* __restrict__ kfull,
                              bf16* __restrict__ vt) {
    int i = blockIdx.x * blockDim.x + threadIdx.x;
    if (i >= BATCH * HEADS * PFX * DHEAD) return;
    int d  = i & 63;
    int p  = (i >> 6) & 15;
    int bh = i >> 10;                       // b*12 + h
    int b  = bh / HEADS, h = bh % HEADS;
    size_t kp = ((((size_t)b * 2 + 0) * PFX + p) * HEADS + h) * DHEAD + d;
    size_t vp = ((((size_t)b * 2 + 1) * PFX + p) * HEADS + h) * DHEAD + d;
    kfull[((size_t)bh * KPAD + p) * DHEAD + d]      = (bf16)prompt[kp];
    vt[((size_t)bh * DHEAD + d) * KPAD + p]         = (bf16)prompt[vp];
    vt[((size_t)bh * DHEAD + d) * KPAD + 592 + p]   = (bf16)0.0f;   // zero pad
}

// ----------------------------- QKV projection ------------------------------
// Y[m,n] = sum_c xb[m,c] * wqkv[n,c]   (m: 9216 tokens, n: 2304)
// 32x32 tile per wave: 2 A-frags x 2 B-frags -> 4 WMMAs per K-step.
// Scatter epilogue into q[B,H,N,D], k_full (+PFX rows), v_t (transposed).
__global__ void qkv_gemm_kernel(const bf16* __restrict__ xb,
                                const bf16* __restrict__ wqkv,
                                bf16* __restrict__ qbuf,
                                bf16* __restrict__ kfull,
                                bf16* __restrict__ vt) {
    const int NT = (3 * CDIM) / 32;                 // 72 col tiles
    int wave = (blockIdx.x * blockDim.x + threadIdx.x) >> 5;
    int lane = threadIdx.x & 31;
    int mt = wave / NT, nt = wave % NT;
    int m0 = mt * 32, n0 = nt * 32;
    int row = lane & 15, hi = lane >> 4;

    const bf16* ap0 = xb   + (size_t)(m0 + row) * CDIM + hi * 8;
    const bf16* ap1 = ap0 + (size_t)16 * CDIM;
    const bf16* bp0 = wqkv + (size_t)(n0 + row) * CDIM + hi * 8;
    const bf16* bp1 = bp0 + (size_t)16 * CDIM;
    v8f cc[2][2] = {};
    for (int k = 0; k < CDIM; k += 32) {
        v16bf a0 = load_frag(ap0 + k), a1 = load_frag(ap1 + k);
        v16bf b0 = load_frag(bp0 + k), b1 = load_frag(bp1 + k);
        cc[0][0] = wmma_bf16(a0, b0, cc[0][0]);
        cc[0][1] = wmma_bf16(a0, b1, cc[0][1]);
        cc[1][0] = wmma_bf16(a1, b0, cc[1][0]);
        cc[1][1] = wmma_bf16(a1, b1, cc[1][1]);
    }

    // D layout: lane -> column (+row) ; VGPR r -> token row (+ r + 8*hi)
    int b = m0 / SEQ;                        // 576 % 32 == 0: uniform per tile
    #pragma unroll
    for (int j = 0; j < 2; ++j) {
        int n     = n0 + 16 * j + row;
        int which = n / CDIM;                // 0=q 1=k 2=v (uniform per tile)
        int c     = n % CDIM;
        int h     = c >> 6, d = c & 63;
        int bh    = b * HEADS + h;
        #pragma unroll
        for (int i = 0; i < 2; ++i) {
            int nn0 = (m0 - b * SEQ) + 16 * i + 8 * hi;
            if (which == 0) {
                bf16* q = qbuf + ((size_t)bh * SEQ + nn0) * DHEAD + d;
                #pragma unroll
                for (int r = 0; r < 8; ++r) q[(size_t)r * DHEAD] = (bf16)cc[i][j][r];
            } else if (which == 1) {
                bf16* kf = kfull + ((size_t)bh * KPAD + PFX + nn0) * DHEAD + d;
                #pragma unroll
                for (int r = 0; r < 8; ++r) kf[(size_t)r * DHEAD] = (bf16)cc[i][j][r];
            } else {
                // v transposed: 8 consecutive tokens -> one 16B store
                bf16 tmp[8];
                #pragma unroll
                for (int r = 0; r < 8; ++r) tmp[r] = (bf16)cc[i][j][r];
                bf16* vp = vt + ((size_t)bh * DHEAD + d) * KPAD + PFX + nn0;
                *reinterpret_cast<uint4*>(vp) = *reinterpret_cast<uint4*>(tmp);
            }
        }
    }
}

// ---------------------- flash attention (transposed) ------------------------
// One wave = 16 queries of one (b,h).  S^T = K_tile(16x64) x Q^T(64x16) so the
// softmax reduction over keys is within-lane + one shfl_xor(16), and the
// probability D-registers feed the P^T B-fragment of O^T += V^T x P^T directly.
__global__ void attn_kernel(const bf16* __restrict__ qbuf,
                            const bf16* __restrict__ kfull,
                            const bf16* __restrict__ vt,
                            const float* __restrict__ act_scale,
                            bf16* __restrict__ obuf) {
    const int QT = SEQ / 16;                        // 36 query tiles
    int wave = (blockIdx.x * blockDim.x + threadIdx.x) >> 5;
    int lane = threadIdx.x & 31;
    int bh = wave / QT, qt = wave % QT;
    int q0 = qt * 16;
    int row = lane & 15, hi = lane >> 4;
    const float scale = 0.125f;                     // 1/sqrt(64)
    float a0 = act_scale[0], a1 = act_scale[1];

    const bf16* qp = qbuf + ((size_t)bh * SEQ + q0 + row) * DHEAD + hi * 8;
    v16bf qb0 = load_frag(qp);                      // d 0..31
    v16bf qb1 = load_frag(qp + 32);                 // d 32..63
    const bf16* kb = kfull + (size_t)bh * KPAD * DHEAD;
    const bf16* vb = vt    + (size_t)bh * DHEAD * KPAD;

    v8f acc0 = {}, acc1 = {}, acc2 = {}, acc3 = {};
    float m = -1e30f, l = 0.0f;

    for (int chunk = 0; chunk < 19; ++chunk) {
        int key0 = chunk * 32;
        bool tail = (chunk == 18);                  // second half is zero-pad

        const bf16* kp0 = kb + (size_t)(key0 + row) * DHEAD + hi * 8;
        v8f s0 = {};
        s0 = wmma_bf16(load_frag(kp0),      qb0, s0);
        s0 = wmma_bf16(load_frag(kp0 + 32), qb1, s0);
        v8f s1 = {};
        if (!tail) {
            const bf16* kp1 = kb + (size_t)(key0 + 16 + row) * DHEAD + hi * 8;
            s1 = wmma_bf16(load_frag(kp1),      qb0, s1);
            s1 = wmma_bf16(load_frag(kp1 + 32), qb1, s1);
        }

        float mloc = -1e30f;
        #pragma unroll
        for (int r = 0; r < 8; ++r) {
            float v = s0[r] * scale;
            if (chunk == 0) v += tanhf(v * a0) * a1;   // gated prefix scores
            s0[r] = v;
            mloc = fmaxf(mloc, v);
        }
        if (!tail) {
            #pragma unroll
            for (int r = 0; r < 8; ++r) {
                float v = s1[r] * scale;
                s1[r] = v;
                mloc = fmaxf(mloc, v);
            }
        }
        mloc = fmaxf(mloc, __shfl_xor(mloc, 16, 32));
        float mnew  = fmaxf(m, mloc);
        float alpha = __expf(m - mnew);
        float rs = 0.0f;
        #pragma unroll
        for (int r = 0; r < 8; ++r) { s0[r] = __expf(s0[r] - mnew); rs += s0[r]; }
        if (!tail) {
            #pragma unroll
            for (int r = 0; r < 8; ++r) { s1[r] = __expf(s1[r] - mnew); rs += s1[r]; }
        }   // tail: s1 stays exactly 0 -> pad keys contribute nothing
        rs += __shfl_xor(rs, 16, 32);
        l = l * alpha + rs;
        m = mnew;
        acc0 *= alpha; acc1 *= alpha; acc2 *= alpha; acc3 *= alpha;

        // pack P^T B-fragment straight from the two score D-tiles
        v16bf pf;
        #pragma unroll
        for (int j = 0; j < 8; ++j) { pf[j] = (bf16)s0[j]; pf[8 + j] = (bf16)s1[j]; }

        const bf16* vp = vb + (size_t)row * KPAD + key0 + hi * 8;
        acc0 = wmma_bf16(load_frag(vp),                  pf, acc0);
        acc1 = wmma_bf16(load_frag(vp + 16 * KPAD),      pf, acc1);
        acc2 = wmma_bf16(load_frag(vp + 32 * KPAD),      pf, acc2);
        acc3 = wmma_bf16(load_frag(vp + 48 * KPAD),      pf, acc3);
    }

    // O^T tile t: lane -> query q0+row ; VGPR r -> d = 16t + r + 8*hi
    float rl = 1.0f / l;
    int b = bh / HEADS, h = bh % HEADS;
    bf16* op = obuf + ((size_t)(b * SEQ + q0 + row)) * CDIM + h * DHEAD + 8 * hi;
    #pragma unroll
    for (int r = 0; r < 8; ++r) {
        op[r]      = (bf16)(acc0[r] * rl);
        op[16 + r] = (bf16)(acc1[r] * rl);
        op[32 + r] = (bf16)(acc2[r] * rl);
        op[48 + r] = (bf16)(acc3[r] * rl);
    }
}

// ----------------------------- output projection ----------------------------
// out[m,n] = sum_c obuf[m,c] * wproj[n,c] + bias[n]   (fp32 out)
// 32x32 tile per wave.
__global__ void proj_gemm_kernel(const bf16* __restrict__ ab,
                                 const bf16* __restrict__ wproj,
                                 const float* __restrict__ bias,
                                 float* __restrict__ out) {
    const int NT = CDIM / 32;                       // 24 col tiles
    int wave = (blockIdx.x * blockDim.x + threadIdx.x) >> 5;
    int lane = threadIdx.x & 31;
    int mt = wave / NT, nt = wave % NT;
    int m0 = mt * 32, n0 = nt * 32;
    int row = lane & 15, hi = lane >> 4;

    const bf16* ap0 = ab    + (size_t)(m0 + row) * CDIM + hi * 8;
    const bf16* ap1 = ap0 + (size_t)16 * CDIM;
    const bf16* bp0 = wproj + (size_t)(n0 + row) * CDIM + hi * 8;
    const bf16* bp1 = bp0 + (size_t)16 * CDIM;
    v8f cc[2][2] = {};
    for (int k = 0; k < CDIM; k += 32) {
        v16bf a0 = load_frag(ap0 + k), a1 = load_frag(ap1 + k);
        v16bf b0 = load_frag(bp0 + k), b1 = load_frag(bp1 + k);
        cc[0][0] = wmma_bf16(a0, b0, cc[0][0]);
        cc[0][1] = wmma_bf16(a0, b1, cc[0][1]);
        cc[1][0] = wmma_bf16(a1, b0, cc[1][0]);
        cc[1][1] = wmma_bf16(a1, b1, cc[1][1]);
    }

    #pragma unroll
    for (int j = 0; j < 2; ++j) {
        int n = n0 + 16 * j + row;
        float bn = bias[n];
        #pragma unroll
        for (int i = 0; i < 2; ++i) {
            float* o = out + (size_t)(m0 + 16 * i + 8 * hi) * CDIM + n;
            #pragma unroll
            for (int r = 0; r < 8; ++r) o[(size_t)r * CDIM] = cc[i][j][r] + bn;
        }
    }
}

// ---------------------------------------------------------------------------
extern "C" void kernel_launch(void* const* d_in, const int* in_sizes, int n_in,
                              void* d_out, int out_size, void* d_ws, size_t ws_size,
                              hipStream_t stream) {
    const float* x         = (const float*)d_in[0];
    const float* prompt    = (const float*)d_in[1];
    const float* act_scale = (const float*)d_in[2];
    const float* qkv_w     = (const float*)d_in[3];
    const float* proj_w    = (const float*)d_in[4];
    const float* proj_b    = (const float*)d_in[5];
    float* out = (float*)d_out;

    char* ws = (char*)d_ws;
    bf16* xb    = (bf16*)ws; ws += (size_t)TOK * CDIM * 2;            // 14.2 MB
    bf16* wqkv  = (bf16*)ws; ws += (size_t)3 * CDIM * CDIM * 2;       //  3.5 MB
    bf16* wproj = (bf16*)ws; ws += (size_t)CDIM * CDIM * 2;           //  1.2 MB
    bf16* qbuf  = (bf16*)ws; ws += (size_t)BATCH * HEADS * SEQ * DHEAD * 2;
    bf16* kfull = (bf16*)ws; ws += (size_t)BATCH * HEADS * KPAD * DHEAD * 2;
    bf16* vt    = (bf16*)ws; ws += (size_t)BATCH * HEADS * DHEAD * KPAD * 2;
    bf16* obuf  = (bf16*)ws; ws += (size_t)TOK * CDIM * 2;

    int n4;
    n4 = TOK * CDIM / 4;
    cast_bf16_kernel<<<(n4 + 255) / 256, 256, 0, stream>>>(x, xb, n4);
    n4 = 3 * CDIM * CDIM / 4;
    cast_bf16_kernel<<<(n4 + 255) / 256, 256, 0, stream>>>(qkv_w, wqkv, n4);
    n4 = CDIM * CDIM / 4;
    cast_bf16_kernel<<<(n4 + 255) / 256, 256, 0, stream>>>(proj_w, wproj, n4);
    int n = BATCH * HEADS * PFX * DHEAD;
    prefix_kernel<<<(n + 255) / 256, 256, 0, stream>>>(prompt, kfull, vt);

    // 288 row-tiles x 72 col-tiles (32x32 each), 8 waves (256 thr) per block
    qkv_gemm_kernel<<<(TOK / 32) * ((3 * CDIM) / 32) / 8, 256, 0, stream>>>(
        xb, wqkv, qbuf, kfull, vt);

    // 192 (b,h) x 36 query tiles
    attn_kernel<<<BATCH * HEADS * (SEQ / 16) / 8, 256, 0, stream>>>(
        qbuf, kfull, vt, act_scale, obuf);

    // 288 row-tiles x 24 col-tiles
    proj_gemm_kernel<<<(TOK / 32) * (CDIM / 32) / 8, 256, 0, stream>>>(
        obuf, wproj, proj_b, out);
}
